// JointAttention_46806553592143
// MI455X (gfx1250) — compile-verified, matrix operand order checked by reading, not run on hardware
//
#include <hip/hip_runtime.h>

// MI455X / gfx1250 implementation of the GQA attention block.
// bf16 V_WMMA_F32_16X16X32_BF16 for all matrix math, f32 accumulation.
// GEMM panels staged global->LDS with CDNA5 async b128 copies (ASYNCcnt),
// V stored transposed so PV WMMA B-fragments are contiguous dword loads,
// 1/sqrt(HD) folded into Q.

#define B_ 2
#define S_ 2048
#define DIM_ 2048
#define H_ 16
#define KVH_ 8
#define HD_ 128

typedef __attribute__((ext_vector_type(16))) __bf16 v16bf;
typedef __attribute__((ext_vector_type(8)))  float  v8f;
typedef __attribute__((ext_vector_type(4)))  int    v4i;

union BF16Frag { v16bf v; unsigned u[8]; };

__device__ __forceinline__ v8f wmma_bf16(v16bf a, v16bf b, v8f c) {
  return __builtin_amdgcn_wmma_f32_16x16x32_bf16(false, a, false, b, (short)0, c, false, false);
}

// K-pair (dword) index held by fragment VGPR v for this lane, per the
// documented 16-bit A-matrix 16x32 layout (B mirrors it with lane = column).
__device__ __forceinline__ int kpat(int v, int lane) {
  return ((v & 4) << 1) + ((lane >> 4) << 2) + (v & 3);
}

__device__ __forceinline__ unsigned short bfbits(float f) {
  union { __bf16 h; unsigned short s; } u; u.h = (__bf16)f; return u.s;
}
__device__ __forceinline__ unsigned pack2(float a, float b) {
  return (unsigned)bfbits(a) | ((unsigned)bfbits(b) << 16);
}

// ---- CDNA5 async global -> LDS 16B copy (per-lane), ASYNCcnt tracked -------
#if defined(__has_builtin)
#if __has_builtin(__builtin_amdgcn_global_load_async_to_lds_b128)
#define HAVE_ASYNC_LDS 1
#endif
#endif

typedef __attribute__((address_space(1))) v4i* gv4i_p;   // global v4i*
typedef __attribute__((address_space(3))) v4i* lv4i_p;   // LDS v4i*

__device__ __forceinline__ void copy16_g2l(unsigned* lds_dst, const unsigned* gsrc) {
#ifdef HAVE_ASYNC_LDS
  __builtin_amdgcn_global_load_async_to_lds_b128(
      (gv4i_p)(unsigned long long)(uintptr_t)gsrc,
      (lv4i_p)(unsigned)(uintptr_t)lds_dst,
      0, 0);
#else
  *(uint4*)lds_dst = *(const uint4*)gsrc;   // b128 load + ds_store_b128 fallback
#endif
}

__device__ __forceinline__ void wait_async_lds() {
#ifdef HAVE_ASYNC_LDS
  asm volatile("s_wait_asynccnt 0x0" ::: "memory");
#endif
}

// ---------------------------------------------------------------- cast f32->bf16
__global__ __launch_bounds__(256) void cast_f32_bf16(const float4* __restrict__ src,
                                                     uint2* __restrict__ dst, int n4) {
  int i = blockIdx.x * 256 + threadIdx.x;
  if (i < n4) {
    float4 v = src[i];
    uint2 r; r.x = pack2(v.x, v.y); r.y = pack2(v.z, v.w);
    dst[i] = r;
  }
}

// ---------------------------------------------------------------- GEMM C = A * B^T
// A: [M][K] bf16 (as dword pairs), B: [N][K] bf16, C: [M][N] f32.
// Block: 256 thr = 8 waves, tile 128(M) x 64(N), K panel 32.
// LDS rows: 36-dword (144B) stride -> 16B-aligned for async b128, conflict-free
// for the 16-lane fragment gathers (36*r mod 64 distinct for r=0..15).
#define LDW 36
__global__ __launch_bounds__(256) void gemm_bf16_nt(const unsigned* __restrict__ A32,
                                                    const unsigned* __restrict__ B32,
                                                    float* __restrict__ C,
                                                    int M, int N, int K) {
  __shared__ __align__(16) unsigned As[128 * LDW];   // 18 KiB
  __shared__ __align__(16) unsigned Bs[64 * LDW];    //  9 KiB
  const int tid = threadIdx.x;
  const int lane = tid & 31, wave = tid >> 5;
  const int waveM = wave >> 1, waveN = wave & 1;
  const int m0 = blockIdx.y * 128, n0 = blockIdx.x * 64;
  const int Kd = K >> 1;
  v8f acc[2][2] = {};

  for (int kb = 0; kb < Kd; kb += 16) {
    __syncthreads();                       // protect LDS before overwrite
    {
      // A panel: 128 rows x 4 x 16B chunks = 512 chunks, 2 per thread
      int c0 = tid, c1 = tid + 256;
      copy16_g2l(&As[(c0 >> 2) * LDW + (c0 & 3) * 4],
                 &A32[(size_t)(m0 + (c0 >> 2)) * Kd + kb + (c0 & 3) * 4]);
      copy16_g2l(&As[(c1 >> 2) * LDW + (c1 & 3) * 4],
                 &A32[(size_t)(m0 + (c1 >> 2)) * Kd + kb + (c1 & 3) * 4]);
      // B panel: 64 rows x 4 chunks = 256 chunks, 1 per thread
      copy16_g2l(&Bs[(tid >> 2) * LDW + (tid & 3) * 4],
                 &B32[(size_t)(n0 + (tid >> 2)) * Kd + kb + (tid & 3) * 4]);
    }
    wait_async_lds();
    __syncthreads();

    BF16Frag a[2], bf[2];
#pragma unroll
    for (int mt = 0; mt < 2; ++mt) {
      int r = waveM * 32 + mt * 16 + (lane & 15);
#pragma unroll
      for (int v = 0; v < 8; ++v) a[mt].u[v] = As[r * LDW + kpat(v, lane)];
    }
#pragma unroll
    for (int nt = 0; nt < 2; ++nt) {
      int r = waveN * 32 + nt * 16 + (lane & 15);
#pragma unroll
      for (int v = 0; v < 8; ++v) bf[nt].u[v] = Bs[r * LDW + kpat(v, lane)];
    }
#pragma unroll
    for (int mt = 0; mt < 2; ++mt)
#pragma unroll
      for (int nt = 0; nt < 2; ++nt)
        acc[mt][nt] = wmma_bf16(a[mt].v, bf[nt].v, acc[mt][nt]);
  }

#pragma unroll
  for (int mt = 0; mt < 2; ++mt)
#pragma unroll
    for (int nt = 0; nt < 2; ++nt)
#pragma unroll
      for (int g = 0; g < 8; ++g) {
        int row = m0 + waveM * 32 + mt * 16 + g + ((lane >> 4) << 3);
        int col = n0 + waveN * 32 + nt * 16 + (lane & 15);
        C[(size_t)row * N + col] = acc[mt][nt][g];
      }
}

// ---------------------------------------------------------------- RMSNorm + RoPE
// qkv f32 [B*S][4096]. One wave per (token, head-slot). Slots: 0-15 q, 16-23 k,
// 24-31 v. Writes q [b,h,s,hd] bf16 (x 1/sqrt(HD)), k [b,kvh,s,hd] bf16,
// v TRANSPOSED [b,kvh,hd,s] bf16.
__global__ __launch_bounds__(256) void norm_rope_kernel(const float* __restrict__ qkv,
                                                        const float* __restrict__ freqs,
                                                        const float* __restrict__ qnw,
                                                        const float* __restrict__ knw,
                                                        unsigned* __restrict__ qo,
                                                        unsigned* __restrict__ ko,
                                                        unsigned short* __restrict__ vo) {
  const int t = blockIdx.x;
  const int b = t >> 11, s = t & 2047;
  const int lane = threadIdx.x & 31, wave = threadIdx.x >> 5;
  const float4* row = (const float4*)(qkv + (size_t)t * 4096);
#pragma unroll
  for (int i = 0; i < 4; ++i) {
    int slot = i * 8 + wave;
    float4 xv = row[slot * 32 + lane];
    if (slot >= 24) {                                   // V: transpose store
      int h = slot - 24, hd = lane * 4;
      size_t hb = (size_t)(b * KVH_ + h) * HD_;
      vo[(hb + hd + 0) * S_ + s] = bfbits(xv.x);
      vo[(hb + hd + 1) * S_ + s] = bfbits(xv.y);
      vo[(hb + hd + 2) * S_ + s] = bfbits(xv.z);
      vo[(hb + hd + 3) * S_ + s] = bfbits(xv.w);
    } else {                                            // Q/K: rmsnorm + rope
      float ss = xv.x * xv.x + xv.y * xv.y + xv.z * xv.z + xv.w * xv.w;
#pragma unroll
      for (int off = 16; off; off >>= 1) ss += __shfl_xor(ss, off, 32);
      float r = rsqrtf(ss * (1.0f / 128.0f) + 1.1920929e-7f);
      const float* nw = (slot < 16) ? qnw : knw;
      float4 w4 = ((const float4*)nw)[lane];
      float y0 = xv.x * r * w4.x, y1 = xv.y * r * w4.y;
      float y2 = xv.z * r * w4.z, y3 = xv.w * r * w4.w;
      // freqs (S,1,64,2,2): [s][p][i][j] at s*256 + p*4 + i*2 + j
      const float* fb = freqs + (size_t)s * 256 + lane * 8;
      float c0 = fb[0], sn0 = fb[2], c1 = fb[4], sn1 = fb[6];
      float o0 = c0 * y0 - sn0 * y1, o1 = sn0 * y0 + c0 * y1;
      float o2 = c1 * y2 - sn1 * y3, o3 = sn1 * y2 + c1 * y3;
      if (slot < 16) {
        const float sc = 0.08838834764831845f;          // HD^-0.5 folded into Q
        o0 *= sc; o1 *= sc; o2 *= sc; o3 *= sc;
        size_t base = ((size_t)(b * H_ + slot) * S_ + s) * 64;
        qo[base + lane * 2]     = pack2(o0, o1);
        qo[base + lane * 2 + 1] = pack2(o2, o3);
      } else {
        size_t base = ((size_t)(b * KVH_ + (slot - 16)) * S_ + s) * 64;
        ko[base + lane * 2]     = pack2(o0, o1);
        ko[base + lane * 2 + 1] = pack2(o2, o3);
      }
    }
  }
}

// ---------------------------------------------------------------- flash attention
// One wave per 16-query tile. Online softmax; P goes through a per-wave padded
// LDS tile (C-layout -> A-layout). V is pre-transposed so B-frags are dword loads.
__global__ __launch_bounds__(256) void flash_attn_kernel(const unsigned* __restrict__ q32,
                                                         const unsigned* __restrict__ k32,
                                                         const unsigned* __restrict__ vt32,
                                                         unsigned short* __restrict__ out16) {
  __shared__ unsigned short Pl[8][16 * 34];              // per-wave 16x32 bf16, pad 2
  const int lane = threadIdx.x & 31, wave = threadIdx.x >> 5;
  const int h = blockIdx.y, b = blockIdx.z;
  const int kvh = h >> 1;                                // NREP = 2, consecutive repeat
  const int q0 = (blockIdx.x * 8 + wave) * 16;

  // Q fragments: 16 x 128 = 4 K=32 A-frags
  BF16Frag qa[4];
  {
    size_t qrow = ((size_t)(b * H_ + h) * S_ + q0 + (lane & 15)) * 64;
#pragma unroll
    for (int c = 0; c < 4; ++c)
#pragma unroll
      for (int v = 0; v < 8; ++v) qa[c].u[v] = q32[qrow + c * 16 + kpat(v, lane)];
  }

  v8f o[8] = {};
  float m8[8], l8[8];
#pragma unroll
  for (int g = 0; g < 8; ++g) { m8[g] = -3.0e38f; l8[g] = 0.0f; }

  for (int kv0 = 0; kv0 < S_; kv0 += 32) {
    // scores: two 16x16 tiles over K=HD=128
    v8f s0 = {}, s1 = {};
    size_t krow0 = ((size_t)(b * KVH_ + kvh) * S_ + kv0 + (lane & 15)) * 64;
    size_t krow1 = krow0 + 16 * 64;
#pragma unroll
    for (int c = 0; c < 4; ++c) {
      BF16Frag kb0, kb1;
#pragma unroll
      for (int v = 0; v < 8; ++v) {
        int kp = c * 16 + kpat(v, lane);
        kb0.u[v] = k32[krow0 + kp];
        kb1.u[v] = k32[krow1 + kp];
      }
      s0 = wmma_bf16(qa[c].v, kb0.v, s0);
      s1 = wmma_bf16(qa[c].v, kb1.v, s1);
    }

    // online softmax row stats (row g + 8*(lane>=16), cols striped over 16 lanes)
    float alpha[8];
#pragma unroll
    for (int g = 0; g < 8; ++g) {
      float mx = fmaxf(s0[g], s1[g]);
#pragma unroll
      for (int off = 8; off; off >>= 1) mx = fmaxf(mx, __shfl_xor(mx, off, 32));
      float mn = fmaxf(m8[g], mx);
      alpha[g] = __expf(m8[g] - mn);
      m8[g] = mn;
      float p0 = __expf(s0[g] - mn);
      float p1 = __expf(s1[g] - mn);
      float rs = p0 + p1;
#pragma unroll
      for (int off = 8; off; off >>= 1) rs += __shfl_xor(rs, off, 32);
      l8[g] = l8[g] * alpha[g] + rs;
      s0[g] = p0; s1[g] = p1;
    }
#pragma unroll
    for (int j = 0; j < 8; ++j)
#pragma unroll
      for (int g = 0; g < 8; ++g) o[j][g] *= alpha[g];

    // P: C-layout -> LDS -> A-layout
    int prow = (lane >> 4) << 3, pcol = lane & 15;
#pragma unroll
    for (int g = 0; g < 8; ++g) {
      Pl[wave][(prow + g) * 34 + pcol]      = bfbits(s0[g]);
      Pl[wave][(prow + g) * 34 + 16 + pcol] = bfbits(s1[g]);
    }
    asm volatile("s_wait_dscnt 0x0" ::: "memory");
    BF16Frag pa;
    int arow = lane & 15;
#pragma unroll
    for (int v = 0; v < 8; ++v)
      pa.u[v] = *(const unsigned*)&Pl[wave][arow * 34 + (kpat(v, lane) << 1)];

    // O += P * V  (V transposed: kv pairs contiguous per hd row)
    size_t vbase = (size_t)(b * KVH_ + kvh) * HD_ * 1024 + (kv0 >> 1);
#pragma unroll
    for (int j = 0; j < 8; ++j) {
      BF16Frag vb;
      int hd = j * 16 + (lane & 15);
#pragma unroll
      for (int v = 0; v < 8; ++v) vb.u[v] = vt32[vbase + (size_t)hd * 1024 + kpat(v, lane)];
      o[j] = wmma_bf16(pa.v, vb.v, o[j]);
    }
  }

  // normalize + store attn output bf16 [B*S][H*HD]
#pragma unroll
  for (int j = 0; j < 8; ++j)
#pragma unroll
    for (int g = 0; g < 8; ++g) {
      int row = q0 + g + ((lane >> 4) << 3);
      int col = h * HD_ + j * 16 + (lane & 15);
      out16[(size_t)(b * S_ + row) * 2048 + col] = bfbits(o[j][g] / l8[g]);
    }
}

// ---------------------------------------------------------------- launch
extern "C" void kernel_launch(void* const* d_in, const int* in_sizes, int n_in,
                              void* d_out, int out_size, void* d_ws, size_t ws_size,
                              hipStream_t stream) {
  const float* x     = (const float*)d_in[0];
  // d_in[1] = x_mask (unused by forward)
  const float* freqs = (const float*)d_in[2];
  const float* w_qkv = (const float*)d_in[3];
  const float* w_out = (const float*)d_in[4];
  const float* q_nw  = (const float*)d_in[5];
  const float* k_nw  = (const float*)d_in[6];
  float* out = (float*)d_out;

  char* ws = (char*)d_ws;
  char* x_bf    = ws;                         // 4096*2048*2 = 16 MiB
  char* wqkv_bf = x_bf    + (size_t)16777216; // 16 MiB
  char* wout_bf = wqkv_bf + (size_t)16777216; //  8 MiB
  char* qkv_f   = wout_bf + (size_t)8388608;  // 4096*4096*4 = 64 MiB
  char* q_bf    = qkv_f   + (size_t)67108864; // 16 MiB
  char* k_bf    = q_bf    + (size_t)16777216; //  8 MiB
  char* vt_bf   = k_bf    + (size_t)8388608;  //  8 MiB
  char* attn_bf = vt_bf   + (size_t)8388608;  // 16 MiB  (total ~152 MiB)

  // 1) f32 -> bf16 casts
  cast_f32_bf16<<<8192, 256, 0, stream>>>((const float4*)x, (uint2*)x_bf, 2097152);
  cast_f32_bf16<<<8192, 256, 0, stream>>>((const float4*)w_qkv, (uint2*)wqkv_bf, 2097152);
  cast_f32_bf16<<<4096, 256, 0, stream>>>((const float4*)w_out, (uint2*)wout_bf, 1048576);

  // 2) qkv = x @ w_qkv^T   (M=4096, N=4096, K=2048)
  gemm_bf16_nt<<<dim3(64, 32), 256, 0, stream>>>((const unsigned*)x_bf,
                                                 (const unsigned*)wqkv_bf,
                                                 (float*)qkv_f, 4096, 4096, 2048);

  // 3) rmsnorm + rope + split/transposed-V
  norm_rope_kernel<<<4096, 256, 0, stream>>>((const float*)qkv_f, freqs, q_nw, k_nw,
                                             (unsigned*)q_bf, (unsigned*)k_bf,
                                             (unsigned short*)vt_bf);

  // 4) flash attention (8 q-tiles/block * 16 blocks, per head, per batch)
  flash_attn_kernel<<<dim3(16, H_, B_), 256, 0, stream>>>((const unsigned*)q_bf,
                                                          (const unsigned*)k_bf,
                                                          (const unsigned*)vt_bf,
                                                          (unsigned short*)attn_bf);

  // 5) out = attn @ w_out^T  (M=4096, N=2048, K=2048) -> f32 d_out
  gemm_bf16_nt<<<dim3(32, 32), 256, 0, stream>>>((const unsigned*)attn_bf,
                                                 (const unsigned*)wout_bf,
                                                 out, 4096, 2048, 2048);
}
